// Codebook_87101936763289
// MI455X (gfx1250) — matrix operand order repaired and checked by reading, not compiled, not run on hardware
//
#include <hip/hip_runtime.h>
#include <hip/hip_bf16.h>
#include <math.h>

typedef __attribute__((ext_vector_type(16))) __bf16 v16bf;
typedef __attribute__((ext_vector_type(8)))  __bf16 v8bf;
typedef __attribute__((ext_vector_type(8)))  float  v8f;

// Problem constants
#define IMG_N   128
#define IMG_C   16
#define IMG_HW  64
#define SEQ     256          // (64/4)^2
#define NS      (IMG_N*SEQ)  // 32768 patches
#define DDIM    256          // 16*4*4
#define KCODES  1024

#define A_STRIDE 264         // 256 + 8 bf16 pad -> conflict-free ds_load_b128
#define CHUNK    64          // codes staged in LDS per double-buffer step
#define NCHUNK   (KCODES/CHUNK)
#define PPB      128         // patches per block (8 waves x 16)

// ---------------------------------------------------------------------------
// Kernel 1: codebook fp32 -> bf16, plus per-code squared norms
// ---------------------------------------------------------------------------
__global__ __launch_bounds__(256) void som_prep(const float* __restrict__ cb,
                                                __bf16* __restrict__ cbh,
                                                float* __restrict__ cnorm) {
    __shared__ float red[256];
    const int b = blockIdx.x;
    const int d = threadIdx.x;
    float v = cb[b * DDIM + d];
    cbh[b * DDIM + d] = (__bf16)v;
    red[d] = v * v;
    __syncthreads();
    for (int s = 128; s > 0; s >>= 1) {
        if (d < s) red[d] += red[d + s];
        __syncthreads();
    }
    if (d == 0) cnorm[b] = red[0];
}

// ---------------------------------------------------------------------------
// Kernel 2: neighbourhood table T[b][:] = sum_k exp(-(k-b)^2/(2*var))*cb[k][:]
// variance = -(256/(2*ln0.1)) -> 1/(2*var) = -ln(0.1)/256
// ---------------------------------------------------------------------------
__global__ __launch_bounds__(256) void som_table(const float* __restrict__ cb,
                                                 float* __restrict__ T) {
    const int b = blockIdx.x;
    const int d = threadIdx.x;
    const float inv2var = 0.0089944727f;
    float acc = 0.0f;
    for (int k = 0; k < KCODES; ++k) {
        float diff = (float)(k - b);
        float wgt  = expf(-diff * diff * inv2var);
        acc = fmaf(wgt, cb[k * DDIM + d], acc);
    }
    T[b * DDIM + d] = acc;
}

// ---------------------------------------------------------------------------
// Kernel 3: BMU via bf16 WMMA. Block = 128 patches (wave w owns patches
// w*16..w*16+15, A-fragments resident in VGPRs). Codebook + code-norms
// streamed through a double-buffered LDS chunk shared by all 8 waves.
// Inner loop: two independent WMMA accumulation chains (2 N-tiles).
// ---------------------------------------------------------------------------
__global__ __launch_bounds__(256) void som_main(const float* __restrict__ x,
                                                const __bf16* __restrict__ cbh,
                                                const float* __restrict__ cnorm,
                                                const float* __restrict__ T,
                                                float* __restrict__ out) {
    __shared__ union {
        __bf16 A[8][16 * A_STRIDE];       // staging only (overlaid)
        __bf16 B[2][CHUNK * A_STRIDE];    // double-buffered code chunk
    } sh;
    __shared__ float cns[2][CHUNK];       // double-buffered code norms
    __shared__ float mval[8][16][16];
    __shared__ int   midx[8][16][16];
    __shared__ int   bmu[PPB];

    const int tid      = threadIdx.x;
    const int w        = tid >> 5;
    const int lane     = tid & 31;
    const int half     = lane >> 4;
    const int cl       = lane & 15;
    const int block_p0 = blockIdx.x * PPB;

    // ---- Stage A (patchify gather, float4 -> 4x bf16): 64 patches x 256 ----
#pragma unroll
    for (int i = 0; i < 16; ++i) {
        int idx = tid + i * 256;               // 0..4095 float4 chunks
        int p   = idx >> 6, q = idx & 63;      // block patch, dim-quad
        int gp  = block_p0 + p;
        int n   = gp >> 8,  s  = gp & 255;
        int PY  = s >> 4,   PX = s & 15;
        int c   = q >> 2,   py = q & 3;        // d = c*16 + py*4 + px
        const float4 v = *(const float4*)
            &x[(((n * IMG_C + c) * IMG_HW) + PY * 4 + py) * IMG_HW + PX * 4];
        __bf16* dst = &sh.A[p >> 4][(p & 15) * A_STRIDE + q * 4];
        dst[0] = (__bf16)v.x; dst[1] = (__bf16)v.y;
        dst[2] = (__bf16)v.z; dst[3] = (__bf16)v.w;
    }
    __syncthreads();

    // ---- A fragments -> registers (resident for entire code loop) ----
    v16bf afrag[8];
#pragma unroll
    for (int kk = 0; kk < 8; ++kk) {
        const int ka = kk * 32 + half * 8;     // ISA 16-bit A 16x32 layout
        const v8bf alo = *(const v8bf*)&sh.A[w][cl * A_STRIDE + ka];
        const v8bf ahi = *(const v8bf*)&sh.A[w][cl * A_STRIDE + ka + 16];
        v16bf a;
#pragma unroll
        for (int e = 0; e < 8; ++e) { a[e] = alo[e]; a[e + 8] = ahi[e]; }
        afrag[kk] = a;
    }
    __syncthreads();   // all waves done reading sh.A before overlay reuse

    // ---- Stage code chunk 0 (+ norms) ----
    const int srow  = tid >> 2;                // 0..63  (code row in chunk)
    const int squad = tid & 3;                 // 64-bf16 quarter-row
    {
        v8bf st[8];
#pragma unroll
        for (int s2 = 0; s2 < 8; ++s2)
            st[s2] = *(const v8bf*)&cbh[srow * DDIM + squad * 64 + s2 * 8];
        __bf16* dst = &sh.B[0][srow * A_STRIDE + squad * 64];
#pragma unroll
        for (int s2 = 0; s2 < 8; ++s2) *(v8bf*)(dst + s2 * 8) = st[s2];
        if (tid < CHUNK) cns[0][tid] = cnorm[tid];
    }
    __syncthreads();

    float best[8];
    int   bidx[8];
#pragma unroll
    for (int r = 0; r < 8; ++r) { best[r] = 3.4e38f; bidx[r] = 0; }

    // ---- Main loop over code chunks (double buffered) ----
    for (int c = 0; c < NCHUNK; ++c) {
        const int cur  = c & 1;
        const bool more = (c + 1 < NCHUNK);

        // issue next chunk's global loads early; WMMA hides the latency.
        // (No other LOADcnt users in this iteration -> full overlap.)
        v8bf st[8];
        float cnv = 0.0f;
        if (more) {
            const int base = (c + 1) * CHUNK + srow;
#pragma unroll
            for (int s2 = 0; s2 < 8; ++s2)
                st[s2] = *(const v8bf*)&cbh[base * DDIM + squad * 64 + s2 * 8];
            if (tid < CHUNK) cnv = cnorm[(c + 1) * CHUNK + tid];
        }

        for (int jj = 0; jj < 2; ++jj) {       // 2 pairs of N-tiles
            const int row0 = (jj * 2 + 0) * 16 + cl;
            const int row1 = (jj * 2 + 1) * 16 + cl;
            v8f acc0 = {}, acc1 = {};
#pragma unroll
            for (int kk = 0; kk < 8; ++kk) {
                const int kb = kk * 32 + half * 8;    // ISA B 32x16 layout
                const __bf16* bp0 = &sh.B[cur][row0 * A_STRIDE + kb];
                const __bf16* bp1 = &sh.B[cur][row1 * A_STRIDE + kb];
                const v8bf b0lo = *(const v8bf*)bp0;
                const v8bf b0hi = *(const v8bf*)(bp0 + 16);
                const v8bf b1lo = *(const v8bf*)bp1;
                const v8bf b1hi = *(const v8bf*)(bp1 + 16);
                v16bf b0, b1;
#pragma unroll
                for (int e = 0; e < 8; ++e) {
                    b0[e] = b0lo[e]; b0[e + 8] = b0hi[e];
                    b1[e] = b1lo[e]; b1[e + 8] = b1hi[e];
                }
                acc0 = __builtin_amdgcn_wmma_f32_16x16x32_bf16(
                    false, afrag[kk], false, b0, (short)0, acc0, false, false);
                acc1 = __builtin_amdgcn_wmma_f32_16x16x32_bf16(
                    false, afrag[kk], false, b1, (short)0, acc1, false, false);
            }
            const int   code0 = c * CHUNK + jj * 32 + cl;
            const float cn0   = cns[cur][jj * 32 + cl];
            const float cn1   = cns[cur][jj * 32 + 16 + cl];
#pragma unroll
            for (int r = 0; r < 8; ++r) {
                // ||flat||^2 is row-constant -> argmin on ||c||^2 - 2*dot
                float d0 = fmaf(-2.0f, acc0[r], cn0);
                float d1 = fmaf(-2.0f, acc1[r], cn1);
                if (d0 < best[r]) { best[r] = d0; bidx[r] = code0; }
                if (d1 < best[r]) { best[r] = d1; bidx[r] = code0 + 16; }
            }
        }

        if (more) {
            __bf16* dst = &sh.B[cur ^ 1][srow * A_STRIDE + squad * 64];
#pragma unroll
            for (int s2 = 0; s2 < 8; ++s2) *(v8bf*)(dst + s2 * 8) = st[s2];
            if (tid < CHUNK) cns[cur ^ 1][tid] = cnv;
        }
        __syncthreads();
    }

    // ---- per-wave cross-lane argmin via LDS ----
#pragma unroll
    for (int r = 0; r < 8; ++r) {
        const int M = r + half * 8;            // C/D layout: M = r + 8*(lane/16)
        mval[w][M][cl] = best[r];
        midx[w][M][cl] = bidx[r];
    }
    __syncthreads();

    if (tid < PPB) {
        const int w2 = tid >> 4, m = tid & 15;
        float bv = 3.4e38f; int bi = 0x7fffffff;
        for (int c2 = 0; c2 < 16; ++c2) {
            float v = mval[w2][m][c2];
            int  ix = midx[w2][m][c2];
            if (v < bv || (v == bv && ix < bi)) { bv = v; bi = ix; }
        }
        bmu[tid] = bi;
    }
    __syncthreads();

    // ---- gather T[bmu[p]] and scatter via unpatchify map (float4) ----
#pragma unroll
    for (int i = 0; i < 32; ++i) {
        int idx = tid + i * 256;               // 0..8191 float4 chunks
        int p   = idx >> 6, q = idx & 63;
        int gp  = block_p0 + p;
        int n   = gp >> 8,  s  = gp & 255;
        int PY  = s >> 4,   PX = s & 15;
        int c   = q >> 2,   py = q & 3;
        const float4 tv = *(const float4*)&T[bmu[p] * DDIM + q * 4];
        *(float4*)&out[(((n * IMG_C + c) * IMG_HW) + PY * 4 + py) * IMG_HW + PX * 4] = tv;
    }
}

// ---------------------------------------------------------------------------
extern "C" void kernel_launch(void* const* d_in, const int* in_sizes, int n_in,
                              void* d_out, int out_size, void* d_ws, size_t ws_size,
                              hipStream_t stream) {
    const float* x  = (const float*)d_in[0];   // (128,16,64,64) fp32
    const float* cb = (const float*)d_in[1];   // (1024,256) fp32
    float* out = (float*)d_out;                // (128,16,64,64) fp32

    char* ws = (char*)d_ws;
    __bf16* cbh  = (__bf16*)ws;                                     // 512 KB
    float*  cnrm = (float*)(ws + (size_t)KCODES * DDIM * 2);        // 4 KB
    float*  T    = (float*)(ws + (size_t)KCODES * DDIM * 2 + 4096); // 1 MB

    som_prep <<<KCODES, 256, 0, stream>>>(cb, cbh, cnrm);
    som_table<<<KCODES, 256, 0, stream>>>(cb, T);
    som_main <<<NS / PPB, 256, 0, stream>>>(x, cbh, cnrm, T, out);
}
